// FusedSelfAttention_9010841387083
// MI455X (gfx1250) — matrix operand order repaired and checked by reading, not compiled, not run on hardware
//
#include <hip/hip_runtime.h>
#include <hip/hip_bf16.h>
#include <stdint.h>

// ---------------------------------------------------------------------------
// MI455X (gfx1250) fused causal self-attention.
// All GEMMs: v_wmma_f32_16x16x32_bf16 (wave32), async-to-LDS double-buffered
// staging (global_load_async_to_lds_b128 + s_wait_asynccnt).
// B=4 H=16 T=2048 D=1024 hd=64.
// ---------------------------------------------------------------------------

typedef __attribute__((ext_vector_type(16))) __bf16 v16bf;
typedef __attribute__((ext_vector_type(8)))  __bf16 v8bf;
typedef __attribute__((ext_vector_type(8)))  float  v8f;
typedef __attribute__((ext_vector_type(4)))  unsigned int   v4u;
typedef unsigned short u16;
typedef unsigned int   u32;

#define N_HEADS 16
#define D_MODEL 1024
#define SEQ     2048
#define BATCH   4
#define HD      64
#define M_ROWS  (BATCH * SEQ)        // 8192

// float -> bf16 bits, round-to-nearest-even
__device__ __forceinline__ u16 f2bf(float f) {
  u32 x = __float_as_uint(f);
  u32 r = x + 0x7FFFu + ((x >> 16) & 1u);
  return (u16)(r >> 16);
}

__device__ __forceinline__ v16bf cat8(v8bf lo, v8bf hi) {
  return __builtin_shufflevector(lo, hi, 0,1,2,3,4,5,6,7,8,9,10,11,12,13,14,15);
}
// A-operand (16x32 bf16): lane row = l%16; elems 0..7 = K[c0..c0+7],
// elems 8..15 = K[c0+16..c0+23], c0 = (l/16)*8.  p = row_base + c0.
__device__ __forceinline__ v16bf load_afrag(const u16* p) {
  v8bf lo = *(const v8bf*)p;
  v8bf hi = *(const v8bf*)(p + 16);
  return cat8(lo, hi);
}
// B-operand (32x16 bf16) from Bt (row = n, K contiguous): lane col = l%16,
// elems 0..7 = K[c0..c0+7], elems 8..15 = K[c0+16..c0+23], c0 = (l/16)*8.  p = n_row_base + c0.
__device__ __forceinline__ v16bf load_bfrag(const u16* p) {
  v8bf lo = *(const v8bf*)p;
  v8bf hi = *(const v8bf*)(p + 16);
  return cat8(lo, hi);
}
__device__ __forceinline__ v8f wmma_bf16(v16bf a, v16bf b, v8f c) {
  return __builtin_amdgcn_wmma_f32_16x16x32_bf16(false, a, false, b,
                                                 (short)0, c, false, false);
}

// CDNA5 async copy global -> LDS (16B per lane), tracked by ASYNCcnt.
// lds_off = AS3 byte address (= low 32 bits of the generic shared pointer).
__device__ __forceinline__ void async_copy_b128(u32 lds_off, const void* gptr) {
  asm volatile("global_load_async_to_lds_b128 %0, %1, off"
               :
               : "v"(lds_off), "v"(gptr)
               : "memory");
}
__device__ __forceinline__ void wait_async0() {
  asm volatile("s_wait_asynccnt 0" ::: "memory");
}
__device__ __forceinline__ u32 lds_off(const void* p) {
  return (u32)(uintptr_t)p;
}

// ---------------------------------------------------------------------------
// Pack kernels
// ---------------------------------------------------------------------------
__global__ __launch_bounds__(256)
void convert_bf16_kernel(const float* __restrict__ src, u16* __restrict__ dst, int n4) {
  int i = blockIdx.x * 256 + threadIdx.x;
  if (i < n4) {
    float4 v = ((const float4*)src)[i];
    uint2 o;
    o.x = (u32)f2bf(v.x) | ((u32)f2bf(v.y) << 16);
    o.y = (u32)f2bf(v.z) | ((u32)f2bf(v.w) << 16);
    ((uint2*)dst)[i] = o;
  }
}

// dst[C][R] (bf16) = transpose(src[R][C] (f32)); R,C multiples of 32
__global__ __launch_bounds__(256)
void transpose_bf16_kernel(const float* __restrict__ src, u16* __restrict__ dst,
                           int R, int C) {
  __shared__ u16 tile[32][33];
  const int tx = threadIdx.x & 31;
  const int ty = threadIdx.x >> 5;           // 0..7
  const int c0 = blockIdx.x * 32;
  const int r0 = blockIdx.y * 32;
#pragma unroll
  for (int i = 0; i < 4; ++i) {
    int r = ty + i * 8;
    tile[r][tx] = f2bf(src[(size_t)(r0 + r) * C + c0 + tx]);
  }
  __syncthreads();
#pragma unroll
  for (int i = 0; i < 4; ++i) {
    int cc = ty + i * 8;
    dst[(size_t)(c0 + cc) * R + r0 + tx] = tile[tx][cc];
  }
}

// ---------------------------------------------------------------------------
// Shared GEMM body: block tile 128(M) x 256(N), BK=32, double-buffered
// async-to-LDS staging. 8 waves as 2(M) x 4(N); wave tile 64x64 = 4x4 WMMA.
// acc layout: acc[mt][nt], C[m][n] with m = j + 8*hl, n = lr.
// ---------------------------------------------------------------------------
#define GEMM_BODY(A_, Bt_)                                                     \
  __shared__ u16 As[2][128 * 32];                                              \
  __shared__ u16 Bs[2][256 * 32];                                              \
  const int tid = threadIdx.x;                                                 \
  const int lane = tid & 31, lr = lane & 15, hl = lane >> 4;                   \
  const int wid = tid >> 5, wm = wid >> 2, wn = wid & 3;                       \
  const int blockM = blockIdx.x * 128;                                         \
  const int blockN = blockIdx.y * 256;                                         \
  const u16* ag[2]; u32 al[2];                                                 \
  _Pragma("unroll")                                                            \
  for (int i = 0; i < 2; ++i) {                                                \
    int idx = tid + i * 256, row = idx >> 2, c8 = (idx & 3) * 8;               \
    ag[i] = (A_) + (size_t)(blockM + row) * D_MODEL + c8;                      \
    al[i] = lds_off(&As[0][row * 32 + c8]);                                    \
  }                                                                            \
  const u16* bg[4]; u32 bl[4];                                                 \
  _Pragma("unroll")                                                            \
  for (int i = 0; i < 4; ++i) {                                                \
    int idx = tid + i * 256, row = idx >> 2, c8 = (idx & 3) * 8;               \
    bg[i] = (Bt_) + (size_t)(blockN + row) * D_MODEL + c8;                     \
    bl[i] = lds_off(&Bs[0][row * 32 + c8]);                                    \
  }                                                                            \
  auto issue = [&](int buf, int k0) {                                          \
    _Pragma("unroll")                                                          \
    for (int i = 0; i < 2; ++i)                                                \
      async_copy_b128(al[i] + buf * (128 * 32 * 2), ag[i] + k0);               \
    _Pragma("unroll")                                                          \
    for (int i = 0; i < 4; ++i)                                                \
      async_copy_b128(bl[i] + buf * (256 * 32 * 2), bg[i] + k0);               \
  };                                                                           \
  v8f acc[4][4] = {};                                                          \
  issue(0, 0);                                                                 \
  auto step = [&](int buf, int k0) {                                           \
    wait_async0();                                                             \
    __syncthreads();                                                           \
    if (k0 + 32 < D_MODEL) issue(buf ^ 1, k0 + 32);                            \
    v16bf af[4], bfv[4];                                                       \
    _Pragma("unroll")                                                          \
    for (int mt = 0; mt < 4; ++mt)                                             \
      af[mt] = load_afrag(&As[buf][(wm * 64 + mt * 16 + lr) * 32 + hl * 8]);   \
    _Pragma("unroll")                                                          \
    for (int nt = 0; nt < 4; ++nt)                                             \
      bfv[nt] = load_bfrag(&Bs[buf][(wn * 64 + nt * 16 + lr) * 32 + hl * 8]);  \
    _Pragma("unroll")                                                          \
    for (int mt = 0; mt < 4; ++mt)                                             \
      _Pragma("unroll")                                                        \
      for (int nt = 0; nt < 4; ++nt)                                           \
        acc[mt][nt] = wmma_bf16(af[mt], bfv[nt], acc[mt][nt]);                 \
  };                                                                           \
  _Pragma("unroll 1")                                                          \
  for (int k0 = 0; k0 < D_MODEL; k0 += 64) {                                   \
    step(0, k0);                                                               \
    step(1, k0 + 32);                                                          \
  }

// ---------------------------------------------------------------------------
// GEMM1: qkv = xb(8192x1024) @ wqkvT(3072x1024)^T, scatter to Q/K/V^T (bf16)
// ---------------------------------------------------------------------------
__global__ __launch_bounds__(256)
void gemm_qkv_kernel(const u16* __restrict__ A, const u16* __restrict__ Bt,
                     u16* __restrict__ Qw, u16* __restrict__ Kw,
                     u16* __restrict__ Vt) {
  GEMM_BODY(A, Bt)

#pragma unroll
  for (int mt = 0; mt < 4; ++mt) {
    const int r0 = blockM + wm * 64 + mt * 16;
    const int bidx = r0 >> 11;            // / 2048
    const int t0 = r0 & 2047;
#pragma unroll
    for (int nt = 0; nt < 4; ++nt) {
      const int n0 = blockN + wn * 64 + nt * 16;
      const int head = n0 / 192;
      const int rem = n0 - head * 192;
      const int seg = rem >> 6;           // 0=Q 1=K 2=V
      const int d0 = rem & 63;
      const int bh = bidx * N_HEADS + head;
      if (seg == 2) {
        // V transposed: Vt[bh][d][t]; 8 consecutive t per lane -> one b128
        u32 pk[4];
#pragma unroll
        for (int jj = 0; jj < 4; ++jj)
          pk[jj] = (u32)f2bf(acc[mt][nt][2 * jj]) |
                   ((u32)f2bf(acc[mt][nt][2 * jj + 1]) << 16);
        u16* dst = Vt + ((size_t)bh * HD + d0 + lr) * SEQ + t0 + hl * 8;
        v4u pv; pv[0] = pk[0]; pv[1] = pk[1]; pv[2] = pk[2]; pv[3] = pk[3];
        *(v4u*)dst = pv;
      } else {
        u16* dst = (seg == 0 ? Qw : Kw) +
                   ((size_t)bh * SEQ + t0 + 8 * hl) * HD + d0 + lr;
#pragma unroll
        for (int j = 0; j < 8; ++j)
          dst[(size_t)j * HD] = f2bf(acc[mt][nt][j]);
      }
    }
  }
}

// ---------------------------------------------------------------------------
// GEMM2: out(f32) = Ob(8192x1024 bf16) @ woT(1024x1024)^T
// ---------------------------------------------------------------------------
__global__ __launch_bounds__(256)
void gemm_out_kernel(const u16* __restrict__ A, const u16* __restrict__ Bt,
                     float* __restrict__ out) {
  GEMM_BODY(A, Bt)

#pragma unroll
  for (int mt = 0; mt < 4; ++mt) {
    const int r0 = blockM + wm * 64 + mt * 16 + 8 * hl;
#pragma unroll
    for (int nt = 0; nt < 4; ++nt) {
      const int n0 = blockN + wn * 64 + nt * 16 + lr;
#pragma unroll
      for (int j = 0; j < 8; ++j)
        out[(size_t)(r0 + j) * D_MODEL + n0] = acc[mt][nt][j];
    }
  }
}

// ---------------------------------------------------------------------------
// Flash attention: grid (T/128, H, B), 8 waves; wave owns 16 query rows.
// Tk block = 64. scores scale = 1/sqrt(D_MODEL) = 1/32.
// ---------------------------------------------------------------------------
__global__ __launch_bounds__(256)
void attn_kernel(const u16* __restrict__ Q, const u16* __restrict__ K,
                 const u16* __restrict__ Vt, u16* __restrict__ O) {
  __shared__ u16 P_lds[8 * 16 * 64];       // per-wave 16x64 bf16 P tile
  const int tid = threadIdx.x;
  const int lane = tid & 31, lr = lane & 15, hl = lane >> 4;
  const int wid = tid >> 5;
  const int q0 = blockIdx.x * 128 + wid * 16;
  const int h = blockIdx.y;
  const int b = blockIdx.z;
  const size_t bh = (size_t)b * N_HEADS + h;

  const u16* Qb = Q + bh * SEQ * HD;
  const u16* Kb = K + bh * SEQ * HD;
  const u16* Vb = Vt + bh * HD * SEQ;
  u16* Pw = P_lds + wid * 16 * 64;

  // Q fragments (A-operand), kept in registers for the whole loop
  v16bf qa[2];
  {
    const u16* pq = Qb + (size_t)(q0 + lr) * HD + hl * 8;
#pragma unroll
    for (int kk = 0; kk < 2; ++kk) qa[kk] = load_afrag(pq + kk * 32);
  }

  v8f o[4] = {};
  float mi[8], li[8];
#pragma unroll
  for (int j = 0; j < 8; ++j) { mi[j] = -3.0e38f; li[j] = 0.0f; }

  const float scale = 0.03125f;            // 1/sqrt(1024)
  const int nkb = (q0 + 15) / 64 + 1;

  for (int kb = 0; kb < nkb; ++kb) {
    const int t0 = kb * 64;
    // ---- S = Q K^T for 4 tiles of 16 cols ----
    v8f s[4] = {};
#pragma unroll
    for (int c = 0; c < 4; ++c) {
      const u16* pk = Kb + (size_t)(t0 + c * 16 + lr) * HD + hl * 8;
#pragma unroll
      for (int kk = 0; kk < 2; ++kk)
        s[c] = wmma_bf16(qa[kk], load_bfrag(pk + kk * 32), s[c]);
    }
    // ---- scale + causal mask ----
#pragma unroll
    for (int c = 0; c < 4; ++c) {
      const int col = t0 + c * 16 + lr;
#pragma unroll
      for (int j = 0; j < 8; ++j) {
        const int row = q0 + 8 * hl + j;
        float v = s[c][j] * scale;
        s[c][j] = (col > row) ? -3.0e38f : v;
      }
    }
    // ---- online softmax (rows striped: 8 rows/lane, 16-lane reductions) ----
#pragma unroll
    for (int j = 0; j < 8; ++j) {
      float mx = fmaxf(fmaxf(s[0][j], s[1][j]), fmaxf(s[2][j], s[3][j]));
#pragma unroll
      for (int off = 1; off < 16; off <<= 1)
        mx = fmaxf(mx, __shfl_xor(mx, off, 32));
      const float mnew = fmaxf(mi[j], mx);
      const float alpha = __expf(mi[j] - mnew);
      float rs = 0.0f;
#pragma unroll
      for (int c = 0; c < 4; ++c) {
        float p = __expf(s[c][j] - mnew);
        s[c][j] = p;
        rs += p;
      }
#pragma unroll
      for (int off = 1; off < 16; off <<= 1)
        rs += __shfl_xor(rs, off, 32);
      mi[j] = mnew;
      li[j] = li[j] * alpha + rs;
#pragma unroll
      for (int c = 0; c < 4; ++c) o[c][j] *= alpha;
    }
    // ---- C-layout -> A-layout for P via private LDS slice ----
#pragma unroll
    for (int c = 0; c < 4; ++c)
#pragma unroll
      for (int j = 0; j < 8; ++j)
        Pw[(8 * hl + j) * 64 + c * 16 + lr] = f2bf(s[c][j]);
    v16bf pa[2];
    {
      const u16* pp = Pw + lr * 64 + hl * 8;
#pragma unroll
      for (int kk = 0; kk < 2; ++kk) pa[kk] = load_afrag(pp + kk * 32);
    }
    // ---- O += P V  (B-operand rows come from V^T, contiguous in t) ----
#pragma unroll
    for (int c = 0; c < 4; ++c) {
      const u16* pv = Vb + (size_t)(c * 16 + lr) * SEQ + t0 + hl * 8;
#pragma unroll
      for (int kk = 0; kk < 2; ++kk)
        o[c] = wmma_bf16(pa[kk], load_bfrag(pv + kk * 32), o[c]);
    }
  }

  // ---- finalize: O /= l, store as [b][t][h*64+d] bf16 ----
  u16* Ob = O + ((size_t)(b * SEQ + q0)) * D_MODEL + h * HD;
#pragma unroll
  for (int j = 0; j < 8; ++j) {
    const float inv = 1.0f / li[j];
    const int row = 8 * hl + j;
#pragma unroll
    for (int c = 0; c < 4; ++c)
      Ob[(size_t)row * D_MODEL + c * 16 + lr] = f2bf(o[c][j] * inv);
  }
}

// ---------------------------------------------------------------------------
extern "C" void kernel_launch(void* const* d_in, const int* in_sizes, int n_in,
                              void* d_out, int out_size, void* d_ws, size_t ws_size,
                              hipStream_t stream) {
  const float* x = (const float*)d_in[0];
  const float* w_qkv = (const float*)d_in[1];
  const float* w_o = (const float*)d_in[2];
  float* out = (float*)d_out;

  char* ws = (char*)d_ws;
  auto alloc = [&](size_t bytes) -> char* {
    char* p = ws;
    ws += (bytes + 255) & ~(size_t)255;
    return p;
  };
  u16* xb    = (u16*)alloc((size_t)M_ROWS * D_MODEL * 2);       // 16 MB
  u16* wqkvT = (u16*)alloc((size_t)3 * D_MODEL * D_MODEL * 2);  // 6 MB
  u16* woT   = (u16*)alloc((size_t)D_MODEL * D_MODEL * 2);      // 2 MB
  u16* Qw    = (u16*)alloc((size_t)BATCH * N_HEADS * SEQ * HD * 2); // 16 MB
  u16* Kw    = (u16*)alloc((size_t)BATCH * N_HEADS * SEQ * HD * 2); // 16 MB
  u16* Vt    = (u16*)alloc((size_t)BATCH * N_HEADS * HD * SEQ * 2); // 16 MB
  u16* Ob    = (u16*)alloc((size_t)M_ROWS * D_MODEL * 2);       // 16 MB

  // 1) pack inputs to bf16 (weights transposed to Bt layout)
  convert_bf16_kernel<<<(M_ROWS * D_MODEL / 4 + 255) / 256, 256, 0, stream>>>(
      x, xb, M_ROWS * D_MODEL / 4);
  transpose_bf16_kernel<<<dim3(3 * D_MODEL / 32, D_MODEL / 32), 256, 0, stream>>>(
      w_qkv, wqkvT, D_MODEL, 3 * D_MODEL);
  transpose_bf16_kernel<<<dim3(D_MODEL / 32, D_MODEL / 32), 256, 0, stream>>>(
      w_o, woT, D_MODEL, D_MODEL);

  // 2) QKV projection with per-head scatter (V stored transposed)
  gemm_qkv_kernel<<<dim3(M_ROWS / 128, 3 * D_MODEL / 256), 256, 0, stream>>>(
      xb, wqkvT, Qw, Kw, Vt);

  // 3) causal flash attention
  attn_kernel<<<dim3(SEQ / 128, N_HEADS, BATCH), 256, 0, stream>>>(Qw, Kw, Vt, Ob);

  // 4) output projection
  gemm_out_kernel<<<dim3(M_ROWS / 128, D_MODEL / 256), 256, 0, stream>>>(Ob, woT, out);
}
